// TPALSTM_2259152797857
// MI455X (gfx1250) — matrix operand order repaired and checked by reading, not compiled, not run on hardware
//
#include <hip/hip_runtime.h>
#include <hip/hip_bf16.h>
#include <math.h>

// Problem constants (from the reference)
#define B_   2048
#define T_   168
#define H_   128
#define FN_  32
#define HOR_ 24

// Tiling
#define BT    64         // batch rows per workgroup (4 M-tiles of 16)
#define NTH   512        // threads per workgroup: 16 wave32s
#define STR   136        // LDS activation row stride in bf16 elements (16B-aligned rows)
#define GSTR  520        // LDS gate row stride in floats (512 + pad)

typedef __attribute__((ext_vector_type(16))) __bf16          v16bf;
typedef __attribute__((ext_vector_type(16))) unsigned short  v16u16;
typedef __attribute__((ext_vector_type(8)))  float           v8f;

__device__ __forceinline__ unsigned short f2bf(float f) {
    union { float f; unsigned u; } v; v.f = f;
    unsigned u = v.u;
    return (unsigned short)((u + 0x7FFFu + ((u >> 16) & 1u)) >> 16);  // RNE
}
__device__ __forceinline__ float bf2f(unsigned short s) {
    union { unsigned u; float f; } v; v.u = ((unsigned)s) << 16;
    return v.f;
}
__device__ __forceinline__ float sigm(float x) { return 1.0f / (1.0f + __expf(-x)); }

__device__ __forceinline__ void sched_fence() {
#if __has_builtin(__builtin_amdgcn_sched_barrier)
    __builtin_amdgcn_sched_barrier(0);   // nothing may be rescheduled across
#endif
}

// ---------------------------------------------------------------------------
// Kernel 1: pack [W_ih | W_hh] (per layer) as bf16 B-fragments for
// v_wmma_f32_16x16x32_bf16.  Fragment index g = (l<<13)|(nt<<8)|(kt<<5)|lane,
// 16 bf16 per lane (element address in shorts = g<<4).
// ISA B layout: N = lane%16, K = 8*(lane/16) + e for e < 8, 16 + 8*(lane/16) + (e-8) for e >= 8.
// ---------------------------------------------------------------------------
__global__ void pack_weights_kernel(const float* __restrict__ W_ih,
                                    const float* __restrict__ W_hh,
                                    unsigned short* __restrict__ Wcat) {
    int g    = blockIdx.x * blockDim.x + threadIdx.x;   // 16384 total
    int lane = g & 31;
    int kt   = (g >> 5) & 7;
    int nt   = (g >> 8) & 31;
    int l    = (g >> 13) & 1;
    int n    = nt * 16 + (lane & 15);
    int hi   = lane >> 4;
    unsigned short* dst = Wcat + ((size_t)g << 4);
#pragma unroll
    for (int e = 0; e < 16; ++e) {
        int k = kt * 32 + (e < 8 ? 8 * hi + e : 16 + 8 * hi + (e - 8));  // global K in [0,256)
        float wv = (k < H_) ? W_ih[((size_t)l * 4 * H_ + n) * H_ + k]
                            : W_hh[((size_t)l * 4 * H_ + n) * H_ + (k - H_)];
        dst[e] = f2bf(wv);
    }
}

// ---------------------------------------------------------------------------
// Kernel 2: persistent batch-tile LSTM.  32 WGs x 512 threads (16 waves).
// Each WG owns 64 batch rows and runs all 168 steps; h in LDS (bf16),
// c in registers, gates staged through LDS.  Per step per layer:
// M=64, N=512, K=256 GEMM -> each wave: 1 M-tile x 8 N-tiles (acc = 64 VGPRs).
// A sched_barrier pins the "load all 8 B-fragments, then run 8 WMMAs"
// shape so the allocator must keep the fragments in distinct registers and
// the L2 weight stream overlaps the matrix pipe.
// ---------------------------------------------------------------------------
__launch_bounds__(NTH)
__global__ void lstm_tpa_kernel(const float* __restrict__ x,
                                const float* __restrict__ hidden_W,
                                const float* __restrict__ hidden_b,
                                const float* __restrict__ b_ih,
                                const float* __restrict__ b_hh,
                                const unsigned short* __restrict__ Wcat,
                                unsigned short* __restrict__ wsH,    // [T-1][B][H] bf16 = relu(h1_t)
                                float* __restrict__ wsHtt) {         // [B][H] f32 = h1_{T-1}
    extern __shared__ char smem[];
    unsigned short* sh_inp = (unsigned short*)smem;          // BT*STR  (xc_t)
    unsigned short* sh_h0  = sh_inp + BT * STR;              // layer0 hidden
    unsigned short* sh_h1  = sh_h0  + BT * STR;              // layer1 hidden
    float* sh_gates = (float*)(sh_h1 + BT * STR);            // BT*GSTR
    float* sh_bias  = sh_gates + BT * GSTR;                  // 2*512 (b_ih+b_hh)

    const int tid   = threadIdx.x;
    const int lane  = tid & 31;
    const int w     = tid >> 5;
    const int mt    = w & 3;        // M tile (rows mt*16..mt*16+15)
    const int ng    = w >> 2;       // N group (N tiles ng*8..ng*8+7)
    const int m     = lane & 15;
    const int hi    = lane >> 4;
    const int row0  = blockIdx.x * BT;

    // one per-lane base; all B-fragment loads become base + immediate offset
    const unsigned short* laneW = Wcat + ((size_t)lane << 4) + ((size_t)ng << 15);

    for (int i = tid; i < 2 * 4 * H_; i += NTH) sh_bias[i] = b_ih[i] + b_hh[i];
    for (int i = tid; i < BT * STR; i += NTH) { sh_h0[i] = 0; sh_h1[i] = 0; }
    float creg[2][16];
#pragma unroll
    for (int l = 0; l < 2; ++l)
#pragma unroll
        for (int it = 0; it < 16; ++it) creg[l][it] = 0.0f;
    __syncthreads();

    for (int t = 0; t < T_; ++t) {
        // stage xc_t = relu(x[b,t]*hidden_W + hidden_b) as bf16
        for (int idx = tid; idx < BT * H_; idx += NTH) {
            int r = idx >> 7, ch = idx & (H_ - 1);
            float xv = x[(size_t)(row0 + r) * T_ + t];
            float v  = fmaxf(fmaf(xv, hidden_W[ch], hidden_b[ch]), 0.0f);
            sh_inp[r * STR + ch] = f2bf(v);
        }
        __syncthreads();

#pragma unroll
        for (int l = 0; l < 2; ++l) {
            const unsigned short* Ain = (l == 0) ? sh_inp : sh_h0;  // K 0..127
            const unsigned short* Ah  = (l == 0) ? sh_h0  : sh_h1;  // K 128..255

            const v8f vz = {0.f, 0.f, 0.f, 0.f, 0.f, 0.f, 0.f, 0.f};
            v8f acc[8];
#pragma unroll
            for (int i = 0; i < 8; ++i) acc[i] = vz;

#pragma unroll 1
            for (int kt = 0; kt < 8; ++kt) {
                // A-fragment from LDS (ISA A layout: lane half selects K-subblocks)
                const unsigned short* src  = (kt < 4) ? Ain : Ah;
                const int ktl = kt & 3;
                const unsigned short* arow = src + (mt * 16 + m) * STR + ktl * 32 + 8 * hi;
                v16u16 au;
#pragma unroll
                for (int e = 0; e < 8; ++e) au[e] = arow[e];
#pragma unroll
                for (int e = 0; e < 8; ++e) au[8 + e] = arow[16 + e];
                v16bf afrag = __builtin_bit_cast(v16bf, au);

                // base for this (l, kt); fragment loads use immediate offsets
                const unsigned short* wbase = laneW + (l << 17) + (kt << 9);
                if (kt < 7) __builtin_prefetch(wbase + (1 << 9), 0, 1);

                // issue all 8 fragment loads (16-load clause) ...
                v16u16 bu[8];
#pragma unroll
                for (int nt2 = 0; nt2 < 8; ++nt2)
                    bu[nt2] = *(const v16u16*)(wbase + (nt2 << 12));  // nt2*8192 B

                // ... pin the schedule: loads may not sink past this point,
                // so all 8 fragments stay live in distinct registers and the
                // per-WMMA waits become descending loadcnt thresholds.
                sched_fence();

#pragma unroll
                for (int nt2 = 0; nt2 < 8; ++nt2) {
                    v16bf bfrag = __builtin_bit_cast(v16bf, bu[nt2]);
                    acc[nt2] = __builtin_amdgcn_wmma_f32_16x16x32_bf16(
                        false, afrag, false, bfrag, (short)0, acc[nt2], false, false);
                }
            }

            // scatter C tiles to LDS (ISA C/D layout: M = r + 8*hi, N = lane%16)
#pragma unroll
            for (int nt2 = 0; nt2 < 8; ++nt2) {
                const int col = (ng * 8 + nt2) * 16 + m;
#pragma unroll
                for (int r = 0; r < 8; ++r)
                    sh_gates[(mt * 16 + 8 * hi + r) * GSTR + col] = acc[nt2][r];
            }
            __syncthreads();

            // fused LSTM cell (torch gate order i,f,g,o)
            unsigned short* hout = (l == 0) ? sh_h0 : sh_h1;
            const float* bb = sh_bias + l * 4 * H_;
#pragma unroll
            for (int it = 0; it < 16; ++it) {
                const int idx = tid + it * NTH;
                const int r = idx >> 7, j = idx & (H_ - 1);
                const float* gr = sh_gates + r * GSTR;
                float gi = gr[j]          + bb[j];
                float gf = gr[j + H_]     + bb[j + H_];
                float gg = gr[j + 2*H_]   + bb[j + 2*H_];
                float go = gr[j + 3*H_]   + bb[j + 3*H_];
                float c  = sigm(gf) * creg[l][it] + sigm(gi) * tanhf(gg);
                float h  = sigm(go) * tanhf(c);
                creg[l][it] = c;
                hout[r * STR + j] = f2bf(h);
                if (l == 1) {
                    if (t < T_ - 1)
                        wsH[((size_t)t * B_ + row0 + r) * H_ + j] = f2bf(fmaxf(h, 0.0f));
                    else
                        wsHtt[(size_t)(row0 + r) * H_ + j] = h;
                }
            }
            __syncthreads();
        }
    }
}

// ---------------------------------------------------------------------------
// Kernel 3: temporal conv.  convbuf[b][f][k] = sum_t relu(h1_t)[b][k]*conv_W[f][t]
// stored in exactly the raw [FN][H] order so the torch .view(-1,H,FN) is a flat read.
// ---------------------------------------------------------------------------
__global__ void conv_kernel(const unsigned short* __restrict__ wsH,
                            const float* __restrict__ conv_W,
                            const float* __restrict__ conv_b,
                            float* __restrict__ convbuf) {
    extern __shared__ char smem[];
    unsigned short* Hb = (unsigned short*)smem;              // (T-1)*H bf16
    float* cw = (float*)(Hb + (T_ - 1) * H_);                // FN*(T-1) f32
    const int b = blockIdx.x, tid = threadIdx.x;
    for (int i = tid; i < (T_ - 1) * H_; i += 256) {
        int t = i >> 7, k = i & (H_ - 1);
        Hb[i] = wsH[((size_t)t * B_ + b) * H_ + k];
    }
    for (int i = tid; i < FN_ * (T_ - 1); i += 256) cw[i] = conv_W[i];
    __syncthreads();
    for (int o = tid; o < FN_ * H_; o += 256) {
        int f = o >> 7, k = o & (H_ - 1);
        float acc = conv_b[f];
        const float* cwf = cw + f * (T_ - 1);
        for (int t = 0; t < T_ - 1; ++t)
            acc = fmaf(bf2f(Hb[t * H_ + k]), cwf[t], acc);
        convbuf[((size_t)b * FN_ + f) * H_ + k] = acc;
    }
}

// ---------------------------------------------------------------------------
// Kernel 4: attention + output head, one block (128 threads) per batch row.
// ---------------------------------------------------------------------------
__global__ void tail_kernel(const float* __restrict__ wsHtt,
                            const float* __restrict__ convbuf,
                            const float* __restrict__ lin1_W, const float* __restrict__ lin1_b,
                            const float* __restrict__ lin2_W, const float* __restrict__ lin2_b,
                            const float* __restrict__ out_W,  const float* __restrict__ out_b,
                            float* __restrict__ out) {
    __shared__ float s_htt[H_];
    __shared__ float s_cv[H_ * FN_];
    __shared__ float s_w[FN_];
    __shared__ float s_alpha[H_];
    __shared__ float s_v[FN_];
    __shared__ float s_nh[H_];
    const int b = blockIdx.x, tid = threadIdx.x;   // 128 threads
    s_htt[tid] = wsHtt[(size_t)b * H_ + tid];
    for (int i = tid; i < H_ * FN_; i += 128)
        s_cv[i] = fmaxf(convbuf[(size_t)b * H_ * FN_ + i], 0.0f);   // flat view == cv[b,h,f]
    __syncthreads();
    if (tid < FN_) {
        float acc = lin1_b[tid];
        for (int k = 0; k < H_; ++k) acc = fmaf(s_htt[k], lin1_W[tid * H_ + k], acc);
        s_w[tid] = acc;
    }
    __syncthreads();
    {   float acc = 0.f;
        for (int f = 0; f < FN_; ++f) acc = fmaf(s_cv[tid * FN_ + f], s_w[f], acc);
        s_alpha[tid] = sigm(acc); }
    __syncthreads();
    if (tid < FN_) {
        float acc = 0.f;
        for (int hh = 0; hh < H_; ++hh) acc = fmaf(s_alpha[hh], s_cv[hh * FN_ + tid], acc);
        s_v[tid] = acc;
    }
    __syncthreads();
    {   float acc = lin2_b[tid];
        const float* wr = lin2_W + tid * (H_ + FN_);
        for (int k = 0; k < H_; ++k) acc = fmaf(s_htt[k], wr[k], acc);
        for (int f = 0; f < FN_; ++f) acc = fmaf(s_v[f], wr[H_ + f], acc);
        s_nh[tid] = acc; }
    __syncthreads();
    if (tid < HOR_) {
        float acc = out_b[tid];
        for (int j = 0; j < H_; ++j) acc = fmaf(s_nh[j], out_W[tid * H_ + j], acc);
        out[(size_t)b * HOR_ + tid] = acc;
    }
}

// ---------------------------------------------------------------------------
extern "C" void kernel_launch(void* const* d_in, const int* in_sizes, int n_in,
                              void* d_out, int out_size, void* d_ws, size_t ws_size,
                              hipStream_t stream) {
    const float* x        = (const float*)d_in[0];
    const float* hidden_W = (const float*)d_in[1];
    const float* hidden_b = (const float*)d_in[2];
    const float* W_ih     = (const float*)d_in[3];
    const float* W_hh     = (const float*)d_in[4];
    const float* b_ih     = (const float*)d_in[5];
    const float* b_hh     = (const float*)d_in[6];
    const float* conv_W   = (const float*)d_in[7];
    const float* conv_b   = (const float*)d_in[8];
    const float* lin1_W   = (const float*)d_in[9];
    const float* lin1_b   = (const float*)d_in[10];
    const float* lin2_W   = (const float*)d_in[11];
    const float* lin2_b   = (const float*)d_in[12];
    const float* out_W    = (const float*)d_in[13];
    const float* out_b    = (const float*)d_in[14];
    float* out = (float*)d_out;

    char* ws = (char*)d_ws;
    unsigned short* Wcat = (unsigned short*)ws;                         // 512 KB packed weights
    size_t off = (size_t)2 * 32 * 8 * 32 * 16 * sizeof(unsigned short);
    unsigned short* wsH = (unsigned short*)(ws + off);                  // 87.6 MB hidden history
    off += (size_t)(T_ - 1) * B_ * H_ * sizeof(unsigned short);
    float* wsHtt = (float*)(ws + off);                                  // 1 MB
    off += (size_t)B_ * H_ * sizeof(float);
    float* convbuf = (float*)(ws + off);                                // 33.5 MB

    pack_weights_kernel<<<64, 256, 0, stream>>>(W_ih, W_hh, Wcat);

    size_t lds = (size_t)3 * BT * STR * sizeof(unsigned short)
               + (size_t)BT * GSTR * sizeof(float)
               + (size_t)2 * 4 * H_ * sizeof(float);                    // ~185 KB < 320 KB/WGP
    lstm_tpa_kernel<<<B_ / BT, NTH, lds, stream>>>(x, hidden_W, hidden_b,
                                                   b_ih, b_hh, Wcat, wsH, wsHtt);

    size_t clds = (size_t)(T_ - 1) * H_ * sizeof(unsigned short)
                + (size_t)FN_ * (T_ - 1) * sizeof(float);
    conv_kernel<<<B_, 256, clds, stream>>>(wsH, conv_W, conv_b, convbuf);

    tail_kernel<<<B_, 128, 0, stream>>>(wsHtt, convbuf, lin1_W, lin1_b,
                                        lin2_W, lin2_b, out_W, out_b, out);
}
